// GINE_13426067767699
// MI455X (gfx1250) — matrix operand order repaired and checked by reading, not compiled, not run on hardware
//
#include <hip/hip_runtime.h>

#define NNODES 50000
#define NEDGES 800000
#define IN_C 128
#define DIM 256
#define OUT_C 128
#define OUT_STRIDE (2 * DIM + OUT_C) /* 640 */

typedef __attribute__((ext_vector_type(16))) __bf16 v16bf;
typedef __attribute__((ext_vector_type(8)))  float  v8f;

union Frag16 { v16bf v; unsigned short s[16]; };

__device__ __forceinline__ unsigned short f2bf(float f) {
    unsigned int u = __float_as_uint(f);
    u += 0x7FFFu + ((u >> 16) & 1u);   // round-to-nearest-even
    return (unsigned short)(u >> 16);
}

// ---------------------------------------------------------------------------
// Edge kernel: for each 16-edge tile, C[16,FOUT] = bf16WMMA(edge_attr_tile @ We),
// then m = relu(C + be + h_prev[src]) and atomic scatter-add into agg[dst].
// ---------------------------------------------------------------------------
template <int FOUT>
__global__ __launch_bounds__(256) void edge_msg_kernel(
    const float* __restrict__ ea,                 // [E, 128]
    const long long* __restrict__ src,
    const long long* __restrict__ dst,
    const float* __restrict__ hprev, int hstride, int hoff,
    const float* __restrict__ We,                 // [128, FOUT]
    const float* __restrict__ be,                 // [FOUT]
    float* __restrict__ agg,                      // [N, FOUT]
    int ntiles)
{
    constexpr int TPW    = FOUT / 128;  // column tiles per wave (8 waves/block)
    constexpr int KSTEPS = IN_C / 32;   // 4
    __shared__ unsigned short ALDS[16][IN_C + 16];
    __shared__ int sSrc[16], sDst[16];

    const int lane = threadIdx.x & 31;
    const int wave = threadIdx.x >> 5;
    const int coln = lane & 15;
    const int hi16 = lane >> 4;   // 0 or 1

    // Weight B-fragments: loaded once per block, live in registers.
    // B in contiguous K order (32x16): lanes 0-15 hold K=0..15, lanes 16-31 K=16..31.
    Frag16 bfr[TPW][KSTEPS];
#pragma unroll
    for (int t = 0; t < TPW; ++t) {
        const int ntile = wave * TPW + t;
#pragma unroll
        for (int kk = 0; kk < KSTEPS; ++kk)
#pragma unroll
            for (int i = 0; i < 16; ++i) {
                int k = kk * 32 + hi16 * 16 + i;
                bfr[t][kk].s[i] = f2bf(We[(size_t)k * FOUT + ntile * 16 + coln]);
            }
    }

    for (int tile = blockIdx.x; tile < ntiles; tile += gridDim.x) {
        __syncthreads();
        if (threadIdx.x < 16)
            sSrc[threadIdx.x] = (int)src[(size_t)tile * 16 + threadIdx.x];
        else if (threadIdx.x < 32)
            sDst[threadIdx.x - 16] = (int)dst[(size_t)tile * 16 + threadIdx.x - 16];
        // Stage 16x128 f32 edge_attr tile -> bf16 LDS (512 float4, 2/thread)
#pragma unroll
        for (int j = 0; j < 2; ++j) {
            int idx = threadIdx.x * 2 + j;
            int row = idx >> 5;
            int c4  = (idx & 31) * 4;
            const float4 v = *(const float4*)(ea + ((size_t)tile * 16 + row) * IN_C + c4);
            ALDS[row][c4 + 0] = f2bf(v.x);
            ALDS[row][c4 + 1] = f2bf(v.y);
            ALDS[row][c4 + 2] = f2bf(v.z);
            ALDS[row][c4 + 3] = f2bf(v.w);
        }
        int nt = tile + gridDim.x;
        if (nt < ntiles)
            __builtin_prefetch(ea + ((size_t)nt * 16 + (threadIdx.x & 15)) * IN_C, 0, 1);
        __syncthreads();

        // A fragments: contiguous K order matching B. lane<16: K in {0..15}.
        const int mrow = lane & 15;
        const int kh8  = hi16 * 8;
        Frag16 afr[KSTEPS];
#pragma unroll
        for (int kk = 0; kk < KSTEPS; ++kk)
#pragma unroll
            for (int i = 0; i < 16; ++i)
                afr[kk].s[i] = ALDS[mrow][kk * 32 + hi16 * 16 + i];

        // Row indices and 32-bit base offsets, hoisted across the TPW loop.
        // (f32 C/D layout: VGPR r holds M = r + 8*hi16.)
        unsigned hbase[8], abase[8];
#pragma unroll
        for (int r = 0; r < 8; ++r) {
            const int m = hi16 * 8 + r;
            hbase[r] = (unsigned)sSrc[m] * (unsigned)hstride + (unsigned)hoff;
            abase[r] = (unsigned)sDst[m] * (unsigned)FOUT;
        }

#pragma unroll
        for (int t = 0; t < TPW; ++t) {
            const int ntile = wave * TPW + t;
            v8f c = {};
#pragma unroll
            for (int kk = 0; kk < KSTEPS; ++kk)
                c = __builtin_amdgcn_wmma_f32_16x16x32_bf16(
                        false, afr[kk].v, false, bfr[t][kk].v,
                        (short)0, c, false, false);
            const unsigned col = (unsigned)(ntile * 16 + coln);
            const float bb = be[col];
            // Batch the 8 gathers so all loads are in flight before any use.
            float xj[8];
#pragma unroll
            for (int r = 0; r < 8; ++r)
                xj[r] = hprev[hbase[r] + col];
#pragma unroll
            for (int r = 0; r < 8; ++r) {
                float v = fmaxf(c[r] + bb + xj[r], 0.0f);
                atomicAdd(&agg[abase[r] + col], v);
            }
        }
    }
}

// ---------------------------------------------------------------------------
// Node kernel: out = tanh((A1 [+ A2]) @ W + bias), A1 strided (h may live in d_out)
// ---------------------------------------------------------------------------
template <int KDIM, int FOUT, bool ADD>
__global__ __launch_bounds__(256) void node_gemm_tanh_kernel(
    const float* __restrict__ A1, int a1s, int a1off,
    const float* __restrict__ A2,                 // [N, KDIM] (ADD only)
    const float* __restrict__ W,                  // [KDIM, FOUT]
    const float* __restrict__ bias,               // [FOUT] or nullptr
    float* __restrict__ out, int os, int ooff,
    int ntiles)
{
    constexpr int TPW    = FOUT / 128;
    constexpr int KSTEPS = KDIM / 32;
    __shared__ unsigned short ALDS[16][KDIM + 16];

    const int lane = threadIdx.x & 31;
    const int wave = threadIdx.x >> 5;
    const int coln = lane & 15;
    const int hi16 = lane >> 4;

    Frag16 bfr[TPW][KSTEPS];
#pragma unroll
    for (int t = 0; t < TPW; ++t) {
        const int ntile = wave * TPW + t;
#pragma unroll
        for (int kk = 0; kk < KSTEPS; ++kk)
#pragma unroll
            for (int i = 0; i < 16; ++i) {
                int k = kk * 32 + hi16 * 16 + i;
                bfr[t][kk].s[i] = f2bf(W[(size_t)k * FOUT + ntile * 16 + coln]);
            }
    }

    for (int tile = blockIdx.x; tile < ntiles; tile += gridDim.x) {
        __syncthreads();
        for (int idx = threadIdx.x; idx < 16 * KDIM; idx += 256) {
            int m = idx / KDIM, k = idx % KDIM;
            unsigned node = (unsigned)(tile * 16 + m);
            float v = A1[node * (unsigned)a1s + (unsigned)a1off + (unsigned)k];
            if (ADD) v += A2[node * (unsigned)KDIM + (unsigned)k];
            ALDS[m][k] = f2bf(v);
        }
        __syncthreads();

        const int mrow = lane & 15;
        const int kh8  = hi16 * 8;
        Frag16 afr[KSTEPS];
#pragma unroll
        for (int kk = 0; kk < KSTEPS; ++kk)
#pragma unroll
            for (int i = 0; i < 16; ++i)
                afr[kk].s[i] = ALDS[mrow][kk * 32 + hi16 * 16 + i];

        unsigned obase[8];
#pragma unroll
        for (int r = 0; r < 8; ++r) {
            unsigned node = (unsigned)(tile * 16 + hi16 * 8 + r);
            obase[r] = node * (unsigned)os + (unsigned)ooff;
        }

#pragma unroll
        for (int t = 0; t < TPW; ++t) {
            const int ntile = wave * TPW + t;
            v8f c = {};
#pragma unroll
            for (int kk = 0; kk < KSTEPS; ++kk)
                c = __builtin_amdgcn_wmma_f32_16x16x32_bf16(
                        false, afr[kk].v, false, bfr[t][kk].v,
                        (short)0, c, false, false);
            const unsigned col = (unsigned)(ntile * 16 + coln);
            const float bb = bias ? bias[col] : 0.0f;
#pragma unroll
            for (int r = 0; r < 8; ++r)
                out[obase[r] + col] = tanhf(c[r] + bb);
        }
    }
}

// ---------------------------------------------------------------------------
// BatchNorm (training mode): column sum/sumsq -> mu, rsig -> apply
// ---------------------------------------------------------------------------
__global__ __launch_bounds__(256) void colsum_accum_kernel(
    const float* __restrict__ a, int rows, int cols, float* __restrict__ acc)
{
    const int tpc  = 256 / cols;               // threads per column (cols: 128 or 256)
    const int col  = threadIdx.x % cols;
    const int rsub = threadIdx.x / cols;
    float s = 0.0f, s2 = 0.0f;
    for (int r = blockIdx.x * tpc + rsub; r < rows; r += gridDim.x * tpc) {
        float v = a[(unsigned)r * (unsigned)cols + (unsigned)col];
        s += v; s2 += v * v;
    }
    __shared__ float red[2][256];
    red[0][threadIdx.x] = s; red[1][threadIdx.x] = s2;
    __syncthreads();
    if (rsub == 0) {
        for (int j = 1; j < tpc; ++j) { s += red[0][col + j * cols]; s2 += red[1][col + j * cols]; }
        atomicAdd(&acc[col], s);
        atomicAdd(&acc[cols + col], s2);
    }
}

__global__ void colstats_finalize_kernel(const float* __restrict__ acc, int rows, int cols,
                                         float* __restrict__ fin)
{
    int c = blockIdx.x * blockDim.x + threadIdx.x;
    if (c < cols) {
        float inv = 1.0f / (float)rows;
        float mu  = acc[c] * inv;
        float var = acc[cols + c] * inv - mu * mu;
        fin[c]        = mu;
        fin[cols + c] = rsqrtf(var + 1e-5f);
    }
}

__global__ __launch_bounds__(256) void bn_apply_kernel(
    const float* __restrict__ a, int rows, int cols,
    const float* __restrict__ fin,
    const float* __restrict__ g, const float* __restrict__ b,
    float* __restrict__ out, int os, int ooff)
{
    unsigned total = (unsigned)rows * (unsigned)cols;
    for (unsigned idx = blockIdx.x * blockDim.x + threadIdx.x; idx < total;
         idx += gridDim.x * blockDim.x) {
        unsigned c = idx % (unsigned)cols;
        unsigned r = idx / (unsigned)cols;
        out[r * (unsigned)os + (unsigned)ooff + c] =
            g[c] * (a[idx] - fin[c]) * fin[cols + c] + b[c];
    }
}

// ---------------------------------------------------------------------------
extern "C" void kernel_launch(void* const* d_in, const int* in_sizes, int n_in,
                              void* d_out, int out_size, void* d_ws, size_t ws_size,
                              hipStream_t stream)
{
    (void)in_sizes; (void)n_in; (void)out_size; (void)ws_size;
    const float*     x    = (const float*)d_in[0];
    const long long* ei   = (const long long*)d_in[1];
    const float*     ea   = (const float*)d_in[2];
    const float*     bng  = (const float*)d_in[3];
    const float*     bnb  = (const float*)d_in[4];
    const float*     We0  = (const float*)d_in[5];
    const float*     be0  = (const float*)d_in[6];
    const float*     W0   = (const float*)d_in[7];
    const float*     b0   = (const float*)d_in[8];
    const float*     g0   = (const float*)d_in[9];
    const float*     bb0  = (const float*)d_in[10];
    const float*     We1  = (const float*)d_in[11];
    const float*     be1  = (const float*)d_in[12];
    const float*     W1   = (const float*)d_in[13];
    const float*     b1   = (const float*)d_in[14];
    const float*     g1   = (const float*)d_in[15];
    const float*     bb1  = (const float*)d_in[16];
    const float*     Wfc  = (const float*)d_in[17];
    float* out = (float*)d_out;

    const long long* src = ei;
    const long long* dst = ei + NEDGES;

    float* ws    = (float*)d_ws;
    float* h     = ws;  ws += (size_t)NNODES * IN_C;   // BN(x), [N,128]
    float* agg0  = ws;  ws += (size_t)NNODES * IN_C;   // layer-0 scatter target
    float* h1pre = ws;  ws += (size_t)NNODES * DIM;    // tanh(...) pre-BN
    float* agg1  = ws;  ws += (size_t)NNODES * DIM;    // layer-1 scatter target
    float* h2pre = ws;  ws += (size_t)NNODES * DIM;
    float* acc0  = ws;  ws += 2 * IN_C;
    float* fin0  = ws;  ws += 2 * IN_C;
    float* accA  = ws;  ws += 2 * DIM;
    float* finA  = ws;  ws += 2 * DIM;
    float* accB  = ws;  ws += 2 * DIM;
    float* finB  = ws;  ws += 2 * DIM;

    hipMemsetAsync(agg0, 0, (size_t)NNODES * IN_C * sizeof(float), stream);
    hipMemsetAsync(agg1, 0, (size_t)NNODES * DIM * sizeof(float), stream);
    hipMemsetAsync(acc0, 0, 2 * IN_C * sizeof(float), stream);
    hipMemsetAsync(accA, 0, 2 * DIM * sizeof(float), stream);
    hipMemsetAsync(accB, 0, 2 * DIM * sizeof(float), stream);

    const int etiles  = NEDGES / 16;   // 50000
    const int ntilesN = NNODES / 16;   // 3125

    // input BN: h = BN(x)
    colsum_accum_kernel<<<512, 256, 0, stream>>>(x, NNODES, IN_C, acc0);
    colstats_finalize_kernel<<<1, 256, 0, stream>>>(acc0, NNODES, IN_C, fin0);
    bn_apply_kernel<<<2048, 256, 0, stream>>>(x, NNODES, IN_C, fin0, bng, bnb, h, IN_C, 0);

    // layer 0
    edge_msg_kernel<IN_C><<<4096, 256, 0, stream>>>(ea, src, dst, h, IN_C, 0, We0, be0, agg0, etiles);
    node_gemm_tanh_kernel<IN_C, DIM, true><<<2048, 256, 0, stream>>>(
        h, IN_C, 0, agg0, W0, b0, h1pre, DIM, 0, ntilesN);
    colsum_accum_kernel<<<512, 256, 0, stream>>>(h1pre, NNODES, DIM, accA);
    colstats_finalize_kernel<<<1, 256, 0, stream>>>(accA, NNODES, DIM, finA);
    bn_apply_kernel<<<2048, 256, 0, stream>>>(h1pre, NNODES, DIM, finA, g0, bb0, out, OUT_STRIDE, 0);

    // layer 1 (h1 read from d_out cols [0,256))
    edge_msg_kernel<DIM><<<4096, 256, 0, stream>>>(ea, src, dst, out, OUT_STRIDE, 0, We1, be1, agg1, etiles);
    node_gemm_tanh_kernel<DIM, DIM, true><<<2048, 256, 0, stream>>>(
        out, OUT_STRIDE, 0, agg1, W1, b1, h2pre, DIM, 0, ntilesN);
    colsum_accum_kernel<<<512, 256, 0, stream>>>(h2pre, NNODES, DIM, accB);
    colstats_finalize_kernel<<<1, 256, 0, stream>>>(accB, NNODES, DIM, finB);
    bn_apply_kernel<<<2048, 256, 0, stream>>>(h2pre, NNODES, DIM, finB, g1, bb1, out, OUT_STRIDE, DIM);

    // fc head: h3 = tanh(h2 @ Wfc) into cols [512,640)
    node_gemm_tanh_kernel<DIM, OUT_C, false><<<2048, 256, 0, stream>>>(
        out, OUT_STRIDE, DIM, nullptr, Wfc, nullptr, out, OUT_STRIDE, 2 * DIM, ntilesN);
}
